// GraphKANField_87162066305783
// MI455X (gfx1250) — matrix operand order repaired and checked by reading, not compiled, not run on hardware
//
#include <hip/hip_runtime.h>
#include <stdint.h>

// ---------------- constants matching the reference ----------------
#define GRIDSZ 5
#define SPLORD 3
#define NCOEF 8            // GRIDSZ + SPLORD
#define HIDDEN 128
#define EDGE_F 2
#define GAT_SLOPE 0.2f
#define ACT_SLOPE 0.01f

typedef __attribute__((ext_vector_type(16))) __bf16 v16bf;
typedef __attribute__((ext_vector_type(8)))  float  v8f;

union BfFrag { v16bf v; uint4 q[2]; };

__device__ __forceinline__ v8f v8zero() {
  v8f z = {0.f, 0.f, 0.f, 0.f, 0.f, 0.f, 0.f, 0.f};
  return z;
}

// float -> bf16 (round to nearest even), stored as raw u16
__device__ __forceinline__ unsigned short f2bf(float f) {
  unsigned u = __float_as_uint(f);
  unsigned r = u + 0x7FFFu + ((u >> 16) & 1u);
  return (unsigned short)(r >> 16);
}

__device__ __forceinline__ float siluf(float x) { return x / (1.0f + __expf(-x)); }
__device__ __forceinline__ float leakyf(float x, float s) { return x > 0.0f ? x : s * x; }

// order-preserving float <-> uint mapping for atomicMax-based segment max
__device__ __forceinline__ unsigned fflip(float f) {
  unsigned u = __float_as_uint(f);
  return (u & 0x80000000u) ? ~u : (u | 0x80000000u);
}
__device__ __forceinline__ float funflip(unsigned u) {
  u = (u & 0x80000000u) ? (u & 0x7FFFFFFFu) : ~u;
  return __uint_as_float(u);
}

// ---- CDNA5 async global->LDS copy (ASYNCcnt path), per CDNA5 ISA ----
// Each active lane copies 16 bytes: LDS[loff] = MEM[gsrc].
__device__ __forceinline__ void async_copy_b128(const void* gsrc, void* lds_dst) {
  unsigned loff = (unsigned)(size_t)lds_dst;  // low 32 bits of generic addr = LDS offset
  asm volatile("global_load_async_to_lds_b128 %0, %1, off"
               :: "v"(loff), "v"(gsrc)
               : "memory");
}
__device__ __forceinline__ void async_wait0() {
  asm volatile("s_wait_asynccnt 0x0" ::: "memory");
}

// K-sweep of v_wmma_f32_16x16x32_bf16.
// aRow/bRow: per-lane row bases (LDS), contiguous in K.
// Per CDNA5 ISA 16-bit layouts:
//   A frag elems 0..7  = K(half*8 .. +7), elems 8..15 = K(16+half*8 .. +7)
//   B frag elems 0..7  = K(half*8 .. +7), elems 8..15 = K(16+half*8 .. +7)
__device__ __forceinline__ void wmma_sweep(v8f& acc, const unsigned short* aRow,
                                           const unsigned short* bRow, int kn, int half) {
  for (int ks = 0; ks < kn; ks += 32) {
    BfFrag a, b;
    a.q[0] = *(const uint4*)(aRow + ks + half * 8);
    a.q[1] = *(const uint4*)(aRow + ks + 16 + half * 8);
    b.q[0] = *(const uint4*)(bRow + ks + half * 8);
    b.q[1] = *(const uint4*)(bRow + ks + 16 + half * 8);
    acc = __builtin_amdgcn_wmma_f32_16x16x32_bf16(false, a.v, false, b.v,
                                                  (short)0, acc, false, false);
  }
}

// ---------------- weight packing ----------------
// W (R x C, f32, row-major)  ->  Bt (C x Kpad, bf16): Bt[c][k] = W[k][c], zero pad k>=R
__global__ void pack_transpose_bf16(const float* __restrict__ W, int R, int Cc,
                                    unsigned short* __restrict__ Bt, int Kpad) {
  int i = blockIdx.x * blockDim.x + threadIdx.x;
  if (i >= Cc * Kpad) return;
  int c = i / Kpad, k = i - c * Kpad;
  float v = (k < R) ? W[(size_t)k * Cc + c] : 0.0f;
  Bt[i] = f2bf(v);
}

// W (R x C f32, already B^T layout) -> Bt (Npad x Kpad bf16), zero pad
__global__ void pack_cast_bf16(const float* __restrict__ W, int R, int Cc,
                               unsigned short* __restrict__ Bt, int Npad, int Kpad) {
  int i = blockIdx.x * blockDim.x + threadIdx.x;
  if (i >= Npad * Kpad) return;
  int n = i / Kpad, k = i - n * Kpad;
  float v = (n < R && k < Cc) ? W[(size_t)n * Cc + k] : 0.0f;
  Bt[i] = f2bf(v);
}

// spline_w (OUT x IN x NCOEF) * scaler (OUT x IN) -> Bt (Npad x IN*NCOEF bf16)
__global__ void pack_spline_bf16(const float* __restrict__ S, const float* __restrict__ Sc,
                                 int OUT, int IN, unsigned short* __restrict__ Bt, int Npad) {
  int i = blockIdx.x * blockDim.x + threadIdx.x;
  int span = IN * NCOEF;
  if (i >= Npad * span) return;
  int n = i / span, kk = i - n * span;
  int ii = kk / NCOEF, g = kk - ii * NCOEF;
  float v = 0.0f;
  if (n < OUT) v = S[((size_t)n * IN + ii) * NCOEF + g] * Sc[(size_t)n * IN + ii];
  Bt[i] = f2bf(v);
}

// ---------------- generic bf16 WMMA GEMM ----------------
// C[M x Ncols] (f32, ldc)  =  preop(A[M x K] f32) * Bt^T   (Bt: Npad x Kpad bf16)
// PRE: 0 = identity, 1 = SiLU on A elements.
// Block: 256 threads (8 wave32), 32 rows x Npad cols per block.
// B tile staged into LDS via global_load_async_to_lds_b128, overlapped with the
// f32->bf16 A-tile conversion.
template <int PRE>
__global__ __launch_bounds__(256) void gemm_bf16_wmma(
    const float* __restrict__ A, int M, int K, int Kpad,
    const unsigned short* __restrict__ Bt, int Npad,
    float* __restrict__ C, int ldc, int Ncols) {
  __shared__ __align__(16) unsigned short sA[32 * 128];   // 8 KB
  __shared__ __align__(16) unsigned short sB[128 * 128];  // 32 KB
  const int tid = threadIdx.x;
  const int row0 = blockIdx.x * 32;

  // 1) kick off async B-tile copy (raw bf16 bytes, ASYNCcnt)
  {
    int nb = (Npad * Kpad) >> 3;  // b128 transfers
    for (int i = tid; i < nb; i += 256)
      async_copy_b128(Bt + ((size_t)i << 3), &sB[i << 3]);
  }

  // 2) stage A tile (f32 -> bf16, optional SiLU, zero-padded) into LDS
  if (K == Kpad && (K & 3) == 0) {
    int rowv = Kpad >> 2;
    for (int i = tid; i < 32 * rowv; i += 256) {
      int r = i / rowv, k = (i - r * rowv) << 2;
      int grow = row0 + r;
      float4 v = make_float4(0.f, 0.f, 0.f, 0.f);
      if (grow < M) v = *(const float4*)&A[(size_t)grow * K + k];
      if (PRE == 1) {
        v.x = siluf(v.x); v.y = siluf(v.y); v.z = siluf(v.z); v.w = siluf(v.w);
      }
      ushort4 o;
      o.x = f2bf(v.x); o.y = f2bf(v.y); o.z = f2bf(v.z); o.w = f2bf(v.w);
      *(ushort4*)&sA[r * Kpad + k] = o;
    }
  } else {
    for (int i = tid; i < 32 * Kpad; i += 256) {
      int r = i / Kpad, k = i - r * Kpad;
      int grow = row0 + r;
      float v = 0.0f;
      if (grow < M && k < K) {
        v = A[(size_t)grow * K + k];
        if (PRE == 1) v = siluf(v);
      }
      sA[i] = f2bf(v);
    }
  }
  async_wait0();
  __syncthreads();

  const int wave = tid >> 5, lane = tid & 31;
  const int half = lane >> 4, lane16 = lane & 15;
  const int NT = Npad >> 4, totalT = NT * 2;

  for (int t = wave; t < totalT; t += 8) {
    int mt = t & 1, nt = t >> 1;
    v8f acc = v8zero();
    const unsigned short* aRow = &sA[(mt * 16 + lane16) * Kpad];
    const unsigned short* bRow = &sB[(nt * 16 + lane16) * Kpad];
    wmma_sweep(acc, aRow, bRow, Kpad, half);
    int col = nt * 16 + lane16;
    if (col < Ncols) {
#pragma unroll
      for (int r = 0; r < 8; ++r) {
        int grow = row0 + mt * 16 + half * 8 + r;
        if (grow < M) C[(size_t)grow * ldc + col] = acc[r];
      }
    }
  }
}

// ---------------- KAN spline term via WMMA ----------------
// Cox-de Boor basis, uniform grid over [-1,1], G=5, k=3 -> 8 coefficients
__device__ __forceinline__ void bspline8(float x, float out[NCOEF]) {
  const float h = 2.0f / (float)GRIDSZ;
  float b[GRIDSZ + 2 * SPLORD];  // 11
#pragma unroll
  for (int t = 0; t < GRIDSZ + 2 * SPLORD; ++t) {
    float g0 = -1.0f + (float)(t - SPLORD) * h;
    b[t] = (x >= g0 && x < g0 + h) ? 1.0f : 0.0f;
  }
#pragma unroll
  for (int j = 1; j <= SPLORD; ++j) {
    float inv = 1.0f / ((float)j * h);
#pragma unroll
    for (int t = 0; t < GRIDSZ + 2 * SPLORD - j; ++t) {
      float gt = -1.0f + (float)(t - SPLORD) * h;
      float gtj = gt + (float)(j + 1) * h;
      b[t] = ((x - gt) * b[t] + (gtj - x) * b[t + 1]) * inv;
    }
  }
#pragma unroll
  for (int g = 0; g < NCOEF; ++g) out[g] = b[g];
}

// C[M x Ncols] += basis(H[M x IN]) * Spl^T    (Spl: Npad x IN*NCOEF bf16)
// K processed in chunks of 16 inputs (= 128 bf16 K-columns). Per chunk, the
// 128x128 B submatrix is async-staged into LDS while the basis is computed.
__global__ __launch_bounds__(256) void kan_spline_wmma(
    const float* __restrict__ H, int M, int IN,
    const unsigned short* __restrict__ Spl, int Npad,
    float* __restrict__ C, int ldc, int Ncols) {
  const int CI = 16;                // inputs per chunk
  const int CHK = CI * NCOEF;       // 128 K-cols per chunk
  __shared__ __align__(16) unsigned short sA[32 * CHK];    // 8 KB
  __shared__ __align__(16) unsigned short sB[128 * CHK];   // 32 KB
  const int tid = threadIdx.x;
  const int row0 = blockIdx.x * 32;
  const int wave = tid >> 5, lane = tid & 31;
  const int half = lane >> 4, lane16 = lane & 15;
  const int NT = Npad >> 4, totalT = NT * 2;
  const int Ktot = IN * NCOEF;
  const int t0 = wave, t1 = wave + 8;

  v8f acc0 = v8zero(), acc1 = v8zero();

  for (int ic = 0; ic < IN; ic += CI) {
    // async-stage B chunk: Npad rows x CHK cols (row stride Ktot in global)
    {
      int nb = (Npad * CHK) >> 3;  // b128 transfers (16 per row)
      for (int i = tid; i < nb; i += 256) {
        int n = i >> 4, seg = i & 15;
        const unsigned short* g = Spl + (size_t)n * Ktot + (size_t)ic * NCOEF + seg * 8;
        async_copy_b128(g, &sB[n * CHK + seg * 8]);
      }
    }

    // stage basis chunk: 32 rows x CI inputs x 8 coeffs
    for (int pp = tid; pp < 32 * CI; pp += 256) {
      int r = pp >> 4, i = pp & (CI - 1);
      int grow = row0 + r;
      float xv = (grow < M) ? H[(size_t)grow * IN + ic + i] : 0.0f;
      float bs[NCOEF];
      bspline8(xv, bs);
      unsigned short* dp = &sA[(r * CI + i) * NCOEF];
#pragma unroll
      for (int g = 0; g < NCOEF; ++g) dp[g] = f2bf(bs[g]);
    }
    async_wait0();
    __syncthreads();

    if (t0 < totalT) {
      int mt = t0 & 1, nt = t0 >> 1;
      const unsigned short* aRow = &sA[(mt * 16 + lane16) * CHK];
      const unsigned short* bRow = &sB[(nt * 16 + lane16) * CHK];
      wmma_sweep(acc0, aRow, bRow, CHK, half);
    }
    if (t1 < totalT) {
      int mt = t1 & 1, nt = t1 >> 1;
      const unsigned short* aRow = &sA[(mt * 16 + lane16) * CHK];
      const unsigned short* bRow = &sB[(nt * 16 + lane16) * CHK];
      wmma_sweep(acc1, aRow, bRow, CHK, half);
    }
    __syncthreads();
  }

  // accumulate into C (base term already stored there)
  if (t0 < totalT) {
    int mt = t0 & 1, nt = t0 >> 1;
    int col = nt * 16 + lane16;
    if (col < Ncols) {
#pragma unroll
      for (int r = 0; r < 8; ++r) {
        int grow = row0 + mt * 16 + half * 8 + r;
        if (grow < M) C[(size_t)grow * ldc + col] += acc0[r];
      }
    }
  }
  if (t1 < totalT) {
    int mt = t1 & 1, nt = t1 >> 1;
    int col = nt * 16 + lane16;
    if (col < Ncols) {
#pragma unroll
      for (int r = 0; r < 8; ++r) {
        int grow = row0 + mt * 16 + half * 8 + r;
        if (grow < M) C[(size_t)grow * ldc + col] += acc1[r];
      }
    }
  }
}

// ---------------- GATv2 edge phase ----------------
__global__ void gat_init(unsigned* __restrict__ nmax, float* __restrict__ den,
                         float* __restrict__ hout, int n) {
  int i = blockIdx.x * blockDim.x + threadIdx.x;
  if (i < n) {
    nmax[i] = 0x007FFFFFu;  // fflip(-inf)
    den[i] = 0.0f;
  }
  if (i < n * HIDDEN) hout[i] = 0.0f;
}

// one wave32 per edge: m = leaky(xl[s]+xr[d]+ea@We); logit = m . att; segment max
__global__ __launch_bounds__(256) void gat_edge_logits(
    const float* __restrict__ xl, const float* __restrict__ xr,
    const int* __restrict__ src, const int* __restrict__ dst,
    const float* __restrict__ ea, const float* __restrict__ We,
    const float* __restrict__ att, float* __restrict__ logits,
    unsigned* __restrict__ nmax, int E) {
  int e = (blockIdx.x * 256 + threadIdx.x) >> 5;
  int lane = threadIdx.x & 31;
  if (e >= E) return;
  int s = src[e], d = dst[e];
  float a0 = ea[2 * e], a1 = ea[2 * e + 1];
  int j0 = lane * 4;
  const float4 vl = *(const float4*)&xl[(size_t)s * HIDDEN + j0];
  const float4 vr = *(const float4*)&xr[(size_t)d * HIDDEN + j0];
  const float4 w0 = *(const float4*)&We[j0];
  const float4 w1 = *(const float4*)&We[HIDDEN + j0];
  const float4 va = *(const float4*)&att[j0];
  float part = 0.0f, m;
  m = leakyf(vl.x + vr.x + a0 * w0.x + a1 * w1.x, GAT_SLOPE); part += m * va.x;
  m = leakyf(vl.y + vr.y + a0 * w0.y + a1 * w1.y, GAT_SLOPE); part += m * va.y;
  m = leakyf(vl.z + vr.z + a0 * w0.z + a1 * w1.z, GAT_SLOPE); part += m * va.z;
  m = leakyf(vl.w + vr.w + a0 * w0.w + a1 * w1.w, GAT_SLOPE); part += m * va.w;
#pragma unroll
  for (int o = 16; o > 0; o >>= 1) part += __shfl_down(part, o, 32);
  if (lane == 0) {
    logits[e] = part;
    atomicMax(&nmax[d], fflip(part));
  }
}

// ex = exp(logit - max[dst]); den[dst] += ex   (ex overwrites logits in place)
__global__ void gat_edge_softmax(float* __restrict__ lg, const int* __restrict__ dst,
                                 const unsigned* __restrict__ nmax,
                                 float* __restrict__ den, int E) {
  int e = blockIdx.x * blockDim.x + threadIdx.x;
  if (e >= E) return;
  int d = dst[e];
  float ex = __expf(lg[e] - funflip(nmax[d]));
  lg[e] = ex;
  atomicAdd(&den[d], ex);
}

// hout[dst] += xl[src] * (ex / den[dst]); one wave32 per edge
__global__ __launch_bounds__(256) void gat_edge_scatter(
    const float* __restrict__ xl, const int* __restrict__ src,
    const int* __restrict__ dst, const float* __restrict__ exv,
    const float* __restrict__ den, float* __restrict__ hout, int E) {
  int e = (blockIdx.x * 256 + threadIdx.x) >> 5;
  int lane = threadIdx.x & 31;
  if (e >= E) return;
  int s = src[e], d = dst[e];
  float alpha = exv[e] / (den[d] + 1e-16f);
  int j0 = lane * 4;
  const float4 vl = *(const float4*)&xl[(size_t)s * HIDDEN + j0];
  float* o = &hout[(size_t)d * HIDDEN + j0];
  atomicAdd(o + 0, vl.x * alpha);
  atomicAdd(o + 1, vl.y * alpha);
  atomicAdd(o + 2, vl.z * alpha);
  atomicAdd(o + 3, vl.w * alpha);
}

// h = [leaky_](h + bias) [+ res]
__global__ void gat_finalize(float* __restrict__ h, const float* __restrict__ bias,
                             const float* __restrict__ res, int n, int act) {
  int i = blockIdx.x * blockDim.x + threadIdx.x;
  if (i >= n * HIDDEN) return;
  float v = h[i] + bias[i & (HIDDEN - 1)];
  if (act) v = leakyf(v, ACT_SLOPE);
  if (res) v += res[i];
  h[i] = v;
}

// ---------------- host orchestration ----------------
extern "C" void kernel_launch(void* const* d_in, const int* in_sizes, int n_in,
                              void* d_out, int out_size, void* d_ws, size_t ws_size,
                              hipStream_t stream) {
  (void)n_in; (void)out_size; (void)ws_size;
  constexpr int NGNN = 15, NKAN = 8;

  const int N = in_sizes[0] / 3;
  const int E = in_sizes[1] / 2;

  int p = 0;
  const float* x  = (const float*)d_in[p++];
  const int*   ei = (const int*)d_in[p++];
  const float* ea = (const float*)d_in[p++];
  const int* src = ei;
  const int* dst = ei + E;

  const float *Wl[NGNN], *Wr[NGNN], *We[NGNN], *Att[NGNN], *Bias[NGNN];
  for (int l = 0; l < NGNN; ++l) {  // dict insertion order: Wl, Wr, We, att, b
    Wl[l]   = (const float*)d_in[p++];
    Wr[l]   = (const float*)d_in[p++];
    We[l]   = (const float*)d_in[p++];
    Att[l]  = (const float*)d_in[p++];
    Bias[l] = (const float*)d_in[p++];
  }
  const float *Kbw[NKAN], *Ksw[NKAN], *Ksc[NKAN];
  for (int j = 0; j < NKAN; ++j) {  // dict insertion order: base_w, spline_w, scaler
    Kbw[j] = (const float*)d_in[p++];
    Ksw[j] = (const float*)d_in[p++];
    Ksc[j] = (const float*)d_in[p++];
  }

  // workspace carve
  uint8_t* w = (uint8_t*)d_ws;
  auto carve = [&](size_t bytes) {
    void* r = (void*)w;
    w += (bytes + 255) & ~(size_t)255;
    return r;
  };
  float* hA = (float*)carve((size_t)N * HIDDEN * 4);
  float* hB = (float*)carve((size_t)N * HIDDEN * 4);
  float* xl = (float*)carve((size_t)N * HIDDEN * 4);
  float* xr = (float*)carve((size_t)N * HIDDEN * 4);
  float* lg = (float*)carve((size_t)E * 4);
  unsigned* nmax = (unsigned*)carve((size_t)N * 4);
  float* den = (float*)carve((size_t)N * 4);
  unsigned short* bWl   = (unsigned short*)carve((size_t)128 * 128 * 2);
  unsigned short* bWr   = (unsigned short*)carve((size_t)128 * 128 * 2);
  unsigned short* bBase = (unsigned short*)carve((size_t)128 * 128 * 2);
  unsigned short* bSpl  = (unsigned short*)carve((size_t)128 * 128 * NCOEF * 2);

  const int MB = (N + 31) / 32;                 // GEMM row blocks
  const int EW = (E * 32 + 255) / 256;          // wave-per-edge blocks
  const int ET = (E + 255) / 256;               // thread-per-edge blocks
  const int NH = (N * HIDDEN + 255) / 256;      // node*feature blocks

  // ---------- GATv2 stack ----------
  float* hcur = nullptr;
  float* hnext = hA;
  for (int l = 0; l < NGNN; ++l) {
    const float* Ain = (l == 0) ? x : hcur;
    int K    = (l == 0) ? 3 : HIDDEN;
    int Kpad = (l == 0) ? 32 : HIDDEN;

    pack_transpose_bf16<<<(128 * Kpad + 255) / 256, 256, 0, stream>>>(Wl[l], K, 128, bWl, Kpad);
    pack_transpose_bf16<<<(128 * Kpad + 255) / 256, 256, 0, stream>>>(Wr[l], K, 128, bWr, Kpad);
    gemm_bf16_wmma<0><<<MB, 256, 0, stream>>>(Ain, N, K, Kpad, bWl, 128, xl, HIDDEN, HIDDEN);
    gemm_bf16_wmma<0><<<MB, 256, 0, stream>>>(Ain, N, K, Kpad, bWr, 128, xr, HIDDEN, HIDDEN);

    gat_init<<<NH, 256, 0, stream>>>(nmax, den, hnext, N);
    gat_edge_logits<<<EW, 256, 0, stream>>>(xl, xr, src, dst, ea, We[l], Att[l], lg, nmax, E);
    gat_edge_softmax<<<ET, 256, 0, stream>>>(lg, dst, nmax, den, E);
    gat_edge_scatter<<<EW, 256, 0, stream>>>(xl, src, dst, lg, den, hnext, E);

    const float* res = (l > 0 && l < NGNN - 1) ? hcur : nullptr;
    int act = (l < NGNN - 1) ? 1 : 0;
    gat_finalize<<<NH, 256, 0, stream>>>(hnext, Bias[l], res, N, act);

    float* tmp = hcur ? hcur : hB;
    hcur = hnext;
    hnext = tmp;
  }

  // ---------- KAN head ----------
  float* cur = hcur;     // 50000 x 128
  float* other = hnext;  // free ping-pong buffer
  for (int j = 0; j < NKAN; ++j) {
    int OUT  = (j < NKAN - 1) ? HIDDEN : 100;
    int Npad = (OUT + 15) & ~15;
    float* Cout = (j < NKAN - 1) ? other : (float*)d_out;

    pack_cast_bf16<<<(Npad * HIDDEN + 255) / 256, 256, 0, stream>>>(
        Kbw[j], OUT, HIDDEN, bBase, Npad, HIDDEN);
    pack_spline_bf16<<<(Npad * HIDDEN * NCOEF + 255) / 256, 256, 0, stream>>>(
        Ksw[j], Ksc[j], OUT, HIDDEN, bSpl, Npad);

    // base term: silu(h) @ base_w^T  (stores)
    gemm_bf16_wmma<1><<<MB, 256, 0, stream>>>(cur, N, HIDDEN, HIDDEN, bBase, Npad,
                                              Cout, OUT, OUT);
    // spline term: B(h) @ (spline_w*scaler)^T  (accumulates)
    kan_spline_wmma<<<MB, 256, 0, stream>>>(cur, N, HIDDEN, bSpl, Npad, Cout, OUT, OUT);

    other = cur;
    cur = Cout;
  }
}
